// MultiHeadAttention_8272107012264
// MI455X (gfx1250) — matrix operand order repaired and checked by reading, not compiled, not run on hardware
//
#include <hip/hip_runtime.h>
#include <hip/hip_bf16.h>

// ---------------------------------------------------------------------------
// MI455X (gfx1250) multi-head attention forward.
//  - all matmuls via v_wmma_f32_16x16x32_f16 (wave32)
//  - f16 operand planes pre-packed once
//  - GEMM W tiles: Tensor Data Mover (tensor_load_to_lds, TENSORcnt)
//  - GEMM A tiles + attention K/V: global_load_async_to_lds_b128 (ASYNCcnt)
//  - double-buffered LDS pipelines throughout
// ---------------------------------------------------------------------------

typedef __attribute__((ext_vector_type(16))) _Float16 v16h;
typedef __attribute__((ext_vector_type(4)))  _Float16 v4h;
typedef __attribute__((ext_vector_type(8)))  float    v8f;
typedef __attribute__((ext_vector_type(4)))  unsigned u32x4;
typedef __attribute__((ext_vector_type(8)))  int      i32x8;
typedef __attribute__((ext_vector_type(4)))  int      i32x4;

#define DM   1024   // d_model
#define SEQ  2048
#define NH   16
#define DK   64

#if __has_builtin(__builtin_amdgcn_tensor_load_to_lds)
#define HAVE_TDM 1
#else
#define HAVE_TDM 0
#endif

// Low 32 bits of a generic pointer to LDS == byte offset within the wave's
// LDS allocation (ISA 10.2: LDS aperture address is truncated to 32 bits).
static __device__ __forceinline__ unsigned lds_lo32(const void* p) {
  return (unsigned)(unsigned long long)p;
}

// One 16-byte async DMA chunk: global -> LDS, tracked by ASYNCcnt.
static __device__ __forceinline__ void async_copy16(unsigned lds_byte, const void* gptr) {
  asm volatile("global_load_async_to_lds_b128 %0, %1, off"
               :: "v"(lds_byte), "v"(gptr) : "memory");
}

#if HAVE_TDM
// TDM 2D tile load: rows x cols f16 tile from a row-major tensor with
// row stride `stride_elems`, landing in LDS with `pad_amount` DWORDs of
// padding every `pad_interval_code` DWORDs (ISA ch.8 D# encoding).
// This toolchain exposes the 6-arg builtin:
//   (u32x4 g0, i32x8 g1, i32x4 g2, i32x4 g3, i32x8 extra, i32 cpol)
static __device__ __forceinline__ void tdm_load_2d_f16(
    unsigned lds_byte, const void* gaddr, unsigned cols, unsigned rows,
    unsigned stride_elems, unsigned pad_interval_code, unsigned pad_amount_code) {
  const unsigned long long ga = (unsigned long long)gaddr;
  u32x4 g0;
  g0[0] = 1u;                                              // count=1 (valid), user mode
  g0[1] = lds_byte;                                        // lds_addr
  g0[2] = (unsigned)(ga & 0xFFFFFFFFu);                    // global_addr[31:0]
  g0[3] = (unsigned)((ga >> 32) & 0x01FFFFFFu) | (2u << 30); // addr[56:32] | type=2
  i32x8 g1;
  g1[0] = (int)((1u << 16)                                 // data_size = 2 bytes
              | (1u << 20)                                 // pad_enable
              | (pad_interval_code << 22)
              | (pad_amount_code << 25));
  g1[1] = (int)((stride_elems & 0xFFFFu) << 16);           // tensor_dim0[15:0]
  g1[2] = (int)(((stride_elems >> 16) & 0xFFFFu)           // tensor_dim0[31:16]
              | ((rows & 0xFFFFu) << 16));                 // tensor_dim1[15:0]
  g1[3] = (int)(((rows >> 16) & 0xFFFFu)                   // tensor_dim1[31:16]
              | (cols << 16));                             // tile_dim0
  g1[4] = (int)rows;                                       // tile_dim1 (tile_dim2=0)
  g1[5] = (int)stride_elems;                               // tensor_dim0_stride[31:0]
  g1[6] = 0;
  g1[7] = 0;
  const i32x4 z4 = {0, 0, 0, 0};
  const i32x8 z8 = {0, 0, 0, 0, 0, 0, 0, 0};
  __builtin_amdgcn_tensor_load_to_lds(g0, g1, z4, z4, z8, 0);
}
#endif

// ---- WMMA fragment loaders (contiguous k = 16*(lane>>4) + e for A and B, wave32) -------------------
static __device__ __forceinline__ v16h load_a_frag(const _Float16* p0, int ld, int k0) {
  const int lane = threadIdx.x & 31;
  const int m    = lane & 15;
  const int kb   = k0 + ((lane >> 4) << 4);
  const _Float16* p = p0 + (size_t)m * ld;
  v16h a;
#pragma unroll
  for (int e = 0; e < 16; ++e) a[e] = p[kb + e];
  return a;
}

static __device__ __forceinline__ v16h load_b_frag(const _Float16* p0, int ld, int k0, int n0) {
  const int lane = threadIdx.x & 31;
  const int n    = n0 + (lane & 15);
  const int kb   = k0 + ((lane >> 4) << 4);
  v16h b;
#pragma unroll
  for (int e = 0; e < 16; ++e) b[e] = p0[(size_t)(kb + e) * ld + n];
  return b;
}

static __device__ __forceinline__ v16h load_bT_frag(const _Float16* p0, int ld, int k0, int n0) {
  const int lane = threadIdx.x & 31;
  const int n    = n0 + (lane & 15);
  const int kb   = k0 + ((lane >> 4) << 4);
  const _Float16* p = p0 + (size_t)n * ld + kb;
  v16h b;
#pragma unroll
  for (int e = 0; e < 16; ++e) b[e] = p[e];
  return b;
}

// ---- fp32 -> f16 pack (one pass) ------------------------------------------
__global__ __launch_bounds__(256) void cvt_f32_f16(const float* __restrict__ in,
                                                   _Float16* __restrict__ out) {
  const size_t i = ((size_t)blockIdx.x * 256 + threadIdx.x) * 4;
  const float4 x = *(const float4*)(in + i);
  v4h h;
  h[0] = (_Float16)x.x; h[1] = (_Float16)x.y;
  h[2] = (_Float16)x.z; h[3] = (_Float16)x.w;
  *(v4h*)(out + i) = h;
}

// ---- GEMM: C[MxN] = A[MxK] * W[KxN] + bias, f16 in / f32 accumulate -------
// 128x128 block tile, BK=64, 8 waves (4x2). W tile staged by the TDM (one
// descriptor per stage, wave 0 only); A tile by per-lane async copies.
template <typename OT>
__global__ __launch_bounds__(256) void wmma_gemm_bias(
    const _Float16* __restrict__ A, const _Float16* __restrict__ W,
    const float* __restrict__ bias, OT* __restrict__ C,
    int M, int N, int K) {
  constexpr int BM = 128, BN = 128, BK = 64;
  constexpr int ALD = BK + 8;    // 72
  constexpr int WLD = BN + 8;    // 136 (272B pitch == TDM pad 4 DWORDs / 64 DWORDs)
  __shared__ _Float16 As[2 * BM * ALD];
  __shared__ _Float16 Ws[2 * BK * WLD];

  const int tid  = threadIdx.x;
  const int wave = tid >> 5;
  const int wm   = wave >> 1;              // 0..3
  const int wn   = wave & 1;               // 0..1
  const int row0 = blockIdx.y * BM;
  const int col0 = blockIdx.x * BN;

  v8f acc[2][4] = {};

  const unsigned asBase = lds_lo32(As);
  const unsigned wsBase = lds_lo32(Ws);

  auto issue_a = [&](int k0, int buf) {
#pragma unroll
    for (int idx = tid; idx < BM * 8; idx += 256) {      // A: 128x64 f16, 16B chunks
      const int r = idx >> 3, c = idx & 7;
      async_copy16(asBase + (unsigned)buf * (BM * ALD * 2) + r * (ALD * 2) + c * 16,
                   A + (size_t)(row0 + r) * K + k0 + c * 8);
    }
  };
  auto issue_w = [&](int k0, int buf) {
#if HAVE_TDM
    if (wave == 0)   // TDM is wave-level (EXEC ignored): issue from one wave
      tdm_load_2d_f16(wsBase + (unsigned)buf * (BK * WLD * 2),
                      W + (size_t)k0 * N + col0,
                      /*cols=*/BN, /*rows=*/BK, /*stride=*/N,
                      /*pad_interval 64 DW=*/5u, /*pad 4 DW=*/3u);
#else
#pragma unroll
    for (int idx = tid; idx < BK * 16; idx += 256) {     // fallback: async chunks
      const int r = idx >> 4, c = idx & 15;
      async_copy16(wsBase + (unsigned)buf * (BK * WLD * 2) + r * (WLD * 2) + c * 16,
                   W + (size_t)(k0 + r) * N + col0 + c * 8);
    }
#endif
  };

  const int stages = K / BK;     // 16
  issue_a(0, 0);
  issue_w(0, 0);
  for (int s = 0; s < stages; ++s) {
    if (s + 1 < stages) {
      issue_a((s + 1) * BK, (s + 1) & 1);                // prefetch next stage
      issue_w((s + 1) * BK, (s + 1) & 1);
#if HAVE_TDM
      asm volatile("s_wait_asynccnt 0x4" ::: "memory");  // stage s's 4 A-chunks done
      if (wave == 0) __builtin_amdgcn_s_wait_tensorcnt((short)1);  // stage-s TDM done
#else
      asm volatile("s_wait_asynccnt 0x8" ::: "memory");
#endif
    } else {
      asm volatile("s_wait_asynccnt 0x0" ::: "memory");
#if HAVE_TDM
      if (wave == 0) __builtin_amdgcn_s_wait_tensorcnt((short)0);
#endif
    }
    __syncthreads();   // releases all waves only after wave0's TENSORcnt wait

    const _Float16* Ab = As + (s & 1) * BM * ALD;
    const _Float16* Wb = Ws + (s & 1) * BK * WLD;
#pragma unroll
    for (int kk = 0; kk < 2; ++kk) {
      v16h af[2], bf[4];
#pragma unroll
      for (int i = 0; i < 2; ++i)
        af[i] = load_a_frag(Ab + (wm * 32 + i * 16) * ALD, ALD, kk * 32);
#pragma unroll
      for (int j = 0; j < 4; ++j)
        bf[j] = load_b_frag(Wb, WLD, kk * 32, wn * 64 + j * 16);
#pragma unroll
      for (int i = 0; i < 2; ++i)
#pragma unroll
        for (int j = 0; j < 4; ++j)
          acc[i][j] = __builtin_amdgcn_wmma_f32_16x16x32_f16(
              false, af[i], false, bf[j], (short)0, acc[i][j], false, false);
    }
    __syncthreads();   // everyone done with buf (s&1) before s+2 lands in it
  }

  // Epilogue: bias + store. C/D layout: col = lane&15, row = v + 8*(lane>>4).
  const int lane = tid & 31;
  const int n    = lane & 15;
  const int half = lane >> 4;
#pragma unroll
  for (int i = 0; i < 2; ++i)
#pragma unroll
    for (int j = 0; j < 4; ++j) {
      const int col = col0 + wn * 64 + j * 16 + n;
      const float bv = bias[col];
#pragma unroll
      for (int v = 0; v < 8; ++v) {
        const int row = row0 + wm * 32 + i * 16 + v + half * 8;
        C[(size_t)row * N + col] = (OT)(acc[i][j][v] + bv);
      }
    }
}

// ---- Flash attention: one block per (b, h, q-tile of 128) -----------------
__global__ __launch_bounds__(256) void wmma_flash_attn(
    const _Float16* __restrict__ Qp, const _Float16* __restrict__ Kp,
    const _Float16* __restrict__ Vp, _Float16* __restrict__ O) {
  constexpr int KLD = DK + 8;   // 72
  constexpr int PLD = 72;
  __shared__ _Float16 Kt[2 * 64 * KLD];
  __shared__ _Float16 Vt[2 * 64 * KLD];
  __shared__ _Float16 Pt[8 * 16 * PLD];   // per-wave P slabs

  const int tid  = threadIdx.x;
  const int wave = tid >> 5;
  const int lane = tid & 31;
  const int n    = lane & 15;
  const int half = lane >> 4;

  const int qt = blockIdx.x & 15;     // q tile (16 tiles of 128)
  const int bh = blockIdx.x >> 4;     // 0..31
  const int b  = bh >> 4;
  const int h  = bh & 15;

  const size_t rowbase = (size_t)b * SEQ;   // row base into [B*S, DM]
  const int hc = h * DK;
  const int qrow0 = qt * 128 + wave * 16;

  const _Float16* Kh = Kp + rowbase * DM + hc;
  const _Float16* Vh = Vp + rowbase * DM + hc;

  // Q fragments for this wave's 16 rows, held in registers the whole loop.
  const _Float16* Qw = Qp + (rowbase + qrow0) * DM + hc;
  v16h aq[2];
  aq[0] = load_a_frag(Qw, DM, 0);
  aq[1] = load_a_frag(Qw, DM, 32);

  float mi[8], li[8];
#pragma unroll
  for (int v = 0; v < 8; ++v) { mi[v] = -1e30f; li[v] = 0.f; }
  v8f o[4] = {};

  _Float16* Pw = Pt + wave * 16 * PLD;
  const unsigned kBase = lds_lo32(Kt);
  const unsigned vBase = lds_lo32(Vt);

  auto issue_kv = [&](int t, int buf) {
#pragma unroll
    for (int idx = tid; idx < 64 * 8; idx += 256) {   // 64x64 f16 = 512 chunks
      const int r = idx >> 3, c = idx & 7;
      const size_t g = (size_t)(t * 64 + r) * DM + c * 8;
      const unsigned l = (unsigned)buf * (64 * KLD * 2) + r * (KLD * 2) + c * 16;
      async_copy16(kBase + l, Kh + g);
      async_copy16(vBase + l, Vh + g);
    }
  };

  constexpr int TSTEPS = SEQ / 64;   // 32
  issue_kv(0, 0);
  for (int t = 0; t < TSTEPS; ++t) {
    if (t + 1 < TSTEPS) {
      issue_kv(t + 1, (t + 1) & 1);
      asm volatile("s_wait_asynccnt 0x4" ::: "memory");  // stage t's 4 chunks done
    } else {
      asm volatile("s_wait_asynccnt 0x0" ::: "memory");
    }
    __syncthreads();
    const _Float16* Ktb = Kt + (t & 1) * 64 * KLD;
    const _Float16* Vtb = Vt + (t & 1) * 64 * KLD;

    // S (16x64) = Q * K^T
    v8f s[4] = {};
#pragma unroll
    for (int kk = 0; kk < 2; ++kk)
#pragma unroll
      for (int j = 0; j < 4; ++j)
        s[j] = __builtin_amdgcn_wmma_f32_16x16x32_f16(
            false, aq[kk], false, load_bT_frag(Ktb, KLD, kk * 32, j * 16),
            (short)0, s[j], false, false);

#pragma unroll
    for (int j = 0; j < 4; ++j)
#pragma unroll
      for (int v = 0; v < 8; ++v) s[j][v] *= 0.125f;   // 1/sqrt(Dk)

    // Row max over 64 kv cols: 4 frags in-lane, then 16-lane shuffle tree.
    float tmax[8];
#pragma unroll
    for (int v = 0; v < 8; ++v)
      tmax[v] = fmaxf(fmaxf(s[0][v], s[1][v]), fmaxf(s[2][v], s[3][v]));
#pragma unroll
    for (int msk = 1; msk <= 8; msk <<= 1)
#pragma unroll
      for (int v = 0; v < 8; ++v)
        tmax[v] = fmaxf(tmax[v], __shfl_xor(tmax[v], msk, 32));

    float alpha[8], rsum[8];
#pragma unroll
    for (int v = 0; v < 8; ++v) {
      const float mn = fmaxf(mi[v], tmax[v]);
      alpha[v] = __expf(mi[v] - mn);
      mi[v] = mn;
      rsum[v] = 0.f;
    }
#pragma unroll
    for (int j = 0; j < 4; ++j)
#pragma unroll
      for (int v = 0; v < 8; ++v) {
        const float p = __expf(s[j][v] - mi[v]);
        s[j][v] = p;
        rsum[v] += p;
      }
#pragma unroll
    for (int msk = 1; msk <= 8; msk <<= 1)
#pragma unroll
      for (int v = 0; v < 8; ++v) rsum[v] += __shfl_xor(rsum[v], msk, 32);
#pragma unroll
    for (int v = 0; v < 8; ++v) li[v] = li[v] * alpha[v] + rsum[v];
#pragma unroll
    for (int j = 0; j < 4; ++j)
#pragma unroll
      for (int v = 0; v < 8; ++v) o[j][v] *= alpha[v];

    // P (C-layout f32) -> per-wave LDS slab as f16 (A-layout source).
#pragma unroll
    for (int j = 0; j < 4; ++j)
#pragma unroll
      for (int v = 0; v < 8; ++v)
        Pw[(v + half * 8) * PLD + j * 16 + n] = (_Float16)s[j][v];
    // Wave-private slab: drain DS queue; lockstep lanes need no barrier.
    asm volatile("s_wait_dscnt 0" ::: "memory");

    // O += P * V
#pragma unroll
    for (int kk = 0; kk < 2; ++kk) {
      const v16h pa = load_a_frag(Pw, PLD, kk * 32);
#pragma unroll
      for (int j = 0; j < 4; ++j)
        o[j] = __builtin_amdgcn_wmma_f32_16x16x32_f16(
            false, pa, false, load_b_frag(Vtb, KLD, kk * 32, j * 16),
            (short)0, o[j], false, false);
    }
    __syncthreads();   // all waves done with buf (t&1) before t+2 lands in it
  }

  // Normalize and write head-concatenated attention output [B*S, DM] (f16).
  _Float16* Ow = O + (rowbase + qrow0) * DM + hc;
#pragma unroll
  for (int v = 0; v < 8; ++v) {
    const float inv = 1.f / li[v];
#pragma unroll
    for (int j = 0; j < 4; ++j)
      Ow[(size_t)(v + half * 8) * DM + j * 16 + n] = (_Float16)(o[j][v] * inv);
  }
}

// ---------------------------------------------------------------------------
extern "C" void kernel_launch(void* const* d_in, const int* in_sizes, int n_in,
                              void* d_out, int out_size, void* d_ws, size_t ws_size,
                              hipStream_t stream) {
  (void)in_sizes; (void)n_in; (void)out_size; (void)ws_size;
  const float* q  = (const float*)d_in[0];
  const float* k  = (const float*)d_in[1];
  const float* v  = (const float*)d_in[2];
  const float* wq = (const float*)d_in[3];
  const float* bq = (const float*)d_in[4];
  const float* wk = (const float*)d_in[5];
  const float* bk = (const float*)d_in[6];
  const float* wv = (const float*)d_in[7];
  const float* bv = (const float*)d_in[8];
  const float* wo = (const float*)d_in[9];
  const float* bo = (const float*)d_in[10];

  const int M = 2 * SEQ;                                    // 4096 rows
  const size_t actE = (size_t)M * DM;                       // 4M elems
  const size_t wgtE = (size_t)DM * DM;                      // 1M elems
  char* ws = (char*)d_ws;
  _Float16* qf  = (_Float16*)(ws);                          // f16 activations
  _Float16* kf  = (_Float16*)(ws + 2 * actE);
  _Float16* vf  = (_Float16*)(ws + 4 * actE);
  _Float16* wqf = (_Float16*)(ws + 6 * actE);               // f16 weights
  _Float16* wkf = (_Float16*)(ws + 6 * actE + 2 * wgtE);
  _Float16* wvf = (_Float16*)(ws + 6 * actE + 4 * wgtE);
  _Float16* wof = (_Float16*)(ws + 6 * actE + 6 * wgtE);
  _Float16* qp  = (_Float16*)(ws + 6 * actE + 8 * wgtE);    // projections
  _Float16* kp  = (_Float16*)(ws + 8 * actE + 8 * wgtE);
  _Float16* vp  = (_Float16*)(ws + 10 * actE + 8 * wgtE);
  _Float16* ao  = (_Float16*)(ws + 12 * actE + 8 * wgtE);   // attn out

  cvt_f32_f16<<<(unsigned)(actE / 4 / 256), 256, 0, stream>>>(q, qf);
  cvt_f32_f16<<<(unsigned)(actE / 4 / 256), 256, 0, stream>>>(k, kf);
  cvt_f32_f16<<<(unsigned)(actE / 4 / 256), 256, 0, stream>>>(v, vf);
  cvt_f32_f16<<<(unsigned)(wgtE / 4 / 256), 256, 0, stream>>>(wq, wqf);
  cvt_f32_f16<<<(unsigned)(wgtE / 4 / 256), 256, 0, stream>>>(wk, wkf);
  cvt_f32_f16<<<(unsigned)(wgtE / 4 / 256), 256, 0, stream>>>(wv, wvf);
  cvt_f32_f16<<<(unsigned)(wgtE / 4 / 256), 256, 0, stream>>>(wo, wof);

  dim3 gg(DM / 128, M / 128);   // (8, 32)
  wmma_gemm_bias<_Float16><<<gg, 256, 0, stream>>>(qf, wqf, bq, qp, M, DM, DM);
  wmma_gemm_bias<_Float16><<<gg, 256, 0, stream>>>(kf, wkf, bk, kp, M, DM, DM);
  wmma_gemm_bias<_Float16><<<gg, 256, 0, stream>>>(vf, wvf, bv, vp, M, DM, DM);

  wmma_flash_attn<<<2 * NH * (SEQ / 128), 256, 0, stream>>>(qp, kp, vp, ao);

  wmma_gemm_bias<float><<<gg, 256, 0, stream>>>(ao, wof, bo, (float*)d_out, M, DM, DM);
}
